// Net_26268019982671
// MI455X (gfx1250) — matrix operand order repaired and checked by reading, not compiled, not run on hardware
//
#include <hip/hip_runtime.h>
#include <stdint.h>

// ---------------------------------------------------------------- types
typedef __bf16 bf16;
typedef __attribute__((ext_vector_type(16))) __bf16 v16bf;
typedef __attribute__((ext_vector_type(8)))  float  v8f;
typedef __attribute__((ext_vector_type(4)))  unsigned int v4u;

union FragBF { v16bf v; v4u q[2]; };

__device__ __forceinline__ bf16 f2b(float f) {
  // round-to-nearest-even f32 -> bf16
  unsigned u = __float_as_uint(f);
  unsigned r = (u + 0x7fffu + ((u >> 16) & 1u)) >> 16;
  unsigned short s = (unsigned short)r;
  bf16 o; __builtin_memcpy(&o, &s, 2); return o;
}
__device__ __forceinline__ float sigf(float x) { return 1.f / (1.f + __expf(-x)); }

// async copy global -> LDS (gfx1250 ASYNCcnt path); GV addressing mode
__device__ __forceinline__ void async_b128(unsigned lds_byte_off, const void* gptr) {
  asm volatile("global_load_async_to_lds_b128 %0, %1, off"
               :: "v"(lds_byte_off), "v"((unsigned long long)(uintptr_t)gptr)
               : "memory");
}

#define NN   64
#define LL   512
#define EE   128
#define DIMH 512
#define DD   1024

// ---------------------------------------------------------------- utility kernels
__global__ void k_cast(const float* __restrict__ in, bf16* __restrict__ out, int n) {
  int i = blockIdx.x * blockDim.x + threadIdx.x;
  if (i < n) out[i] = f2b(in[i]);
}
__global__ void k_addb(const float* __restrict__ a, const float* __restrict__ b,
                       float* __restrict__ o, int n) {
  int i = blockIdx.x * blockDim.x + threadIdx.x;
  if (i < n) o[i] = a[i] + b[i];
}
__global__ void k_zero(float* __restrict__ p, int n) {
  int i = blockIdx.x * blockDim.x + threadIdx.x;
  if (i < n) p[i] = 0.f;
}
__global__ void k_embed(const int* __restrict__ X, const float* __restrict__ W,
                        bf16* __restrict__ out, int total) {
  int i = blockIdx.x * blockDim.x + threadIdx.x;   // over N*L*E
  if (i >= total) return;
  int e = i & (EE - 1);
  int row = i >> 7;                                // E == 128
  out[i] = f2b(W[(size_t)X[row] * EE + e]);
}

// ---------------------------------------------------------------- direct WMMA GEMM
// (used for the small latency-critical GEMMs: recurrent steps, decoder 64-row GEMMs)
// C[M x N] = A0[M x K0] * Bt0[N x K0]^T (+ A1[M x K1] * Bt1[N x K1]^T) (+bias) (tanh?)
__global__ void k_gemm(const bf16* __restrict__ A0, int lda0, int K0,
                       const bf16* __restrict__ A1, int lda1, int K1,
                       const bf16* __restrict__ Bt0, const bf16* __restrict__ Bt1,
                       float* __restrict__ C, int ldc,
                       int Ntiles, int totalTiles,
                       const float* __restrict__ bias, int flags) {
  int wave = (int)((blockIdx.x * blockDim.x + threadIdx.x) >> 5);
  if (wave >= totalTiles) return;                  // whole-wave exit: EXEC stays all-ones
  int lane = threadIdx.x & 31;
  int mt = wave / Ntiles, nt = wave - mt * Ntiles;
  int half = lane >> 4, r = lane & 15;

  v8f acc = {};
  {
    const bf16* arow = A0 + (size_t)(mt * 16 + r) * lda0;
    const bf16* brow = Bt0 + (size_t)(nt * 16 + r) * K0;
    for (int k = 0; k < K0; k += 32) {
      FragBF a, b;
      a.q[0] = *(const v4u*)(arow + k + half * 8);        // A: K k..k+7 / k+8..15
      a.q[1] = *(const v4u*)(arow + k + 16 + half * 8);   //    K k+16..23 / k+24..31
      b.q[0] = *(const v4u*)(brow + k + half * 8);        // B: K k..k+7 / k+8..15
      b.q[1] = *(const v4u*)(brow + k + 16 + half * 8);   //    K k+16..23 / k+24..31
      acc = __builtin_amdgcn_wmma_f32_16x16x32_bf16(false, a.v, false, b.v,
                                                    (short)0, acc, false, false);
    }
  }
  if (A1) {
    const bf16* arow = A1 + (size_t)(mt * 16 + r) * lda1;
    const bf16* brow = Bt1 + (size_t)(nt * 16 + r) * K1;
    for (int k = 0; k < K1; k += 32) {
      FragBF a, b;
      a.q[0] = *(const v4u*)(arow + k + half * 8);
      a.q[1] = *(const v4u*)(arow + k + 16 + half * 8);
      b.q[0] = *(const v4u*)(brow + k + half * 8);
      b.q[1] = *(const v4u*)(brow + k + 16 + half * 8);
      acc = __builtin_amdgcn_wmma_f32_16x16x32_bf16(false, a.v, false, b.v,
                                                    (short)0, acc, false, false);
    }
  }
  float bv = bias ? bias[nt * 16 + r] : 0.f;
  float* crow = C + (size_t)(mt * 16 + half * 8) * ldc + nt * 16 + r;
#pragma unroll
  for (int v = 0; v < 8; ++v) {                    // m = v + 8*half, n = r
    float x = acc[v] + bv;
    if (flags & 1) x = tanhf(x);
    crow[(size_t)v * ldc] = x;
  }
}

// ---------------------------------------------------------------- LDS-staged WMMA GEMM
// Big GEMMs. Block tile 64M x 256N; 8 waves as 2x4 grid; each wave 2Mx4N 16x16 subtiles.
// A/B tiles double-buffered in LDS via global_load_async_to_lds_b128 (ASYNCcnt).
#define STG 10240     // halves per stage: A 64*32 + B 256*32
__global__ void __launch_bounds__(256) k_gemm_lds(
    const bf16* __restrict__ A, int lda, int K,
    const bf16* __restrict__ Bt,
    float* __restrict__ C, int ldc,
    int Nblocks, int Ntotal,
    const float* __restrict__ bias, int flags) {
  __shared__ bf16 smem[2 * STG];                   // 40 KB of the 320 KB WGP LDS

  const int tid  = threadIdx.x;
  const int wv   = tid >> 5;
  const int lane = tid & 31;
  const int wm = wv >> 2, wn = wv & 3;
  const int half = lane >> 4, r = lane & 15;

  const int mb = blockIdx.x / Nblocks;
  const int nb = blockIdx.x - mb * Nblocks;
  const int m0 = mb * 64;
  const int n0 = nb * 256;

  const unsigned smbase = (unsigned)(uintptr_t)(void*)smem;   // LDS byte offset

  // per-thread async-load geometry (constant across K stages)
  const int arow = tid >> 2, aseg = tid & 3;       // A: 64 rows x 4 x 16B
  const bf16* agp = A + (size_t)(m0 + arow) * lda + aseg * 8;
  const unsigned aoff = (unsigned)((arow * 32 + aseg * 8) * 2);

  const bf16* bgp[4];
  unsigned    boff[4];
#pragma unroll
  for (int j = 0; j < 4; ++j) {                    // B: 256 rows x 4 x 16B
    int u = tid + 256 * j;
    int brow = u >> 2, bseg = u & 3;
    int grow = n0 + brow; if (grow > Ntotal - 1) grow = Ntotal - 1;   // clamp (cls N=10000)
    bgp[j]  = Bt + (size_t)grow * K + bseg * 8;
    boff[j] = (unsigned)((2048 + brow * 32 + bseg * 8) * 2);
  }

  v8f acc[2][4] = {};
  const int nk = K >> 5;

  // prologue: stage 0
  {
    async_b128(smbase + aoff, agp);
#pragma unroll
    for (int j = 0; j < 4; ++j) async_b128(smbase + boff[j], bgp[j]);
  }

  for (int ks = 0; ks < nk; ++ks) {
    const int cur = ks & 1;
    if (ks + 1 < nk) {                             // kick next stage, then wait for current
      const unsigned sb = smbase + (unsigned)((cur ^ 1) * STG * 2);
      const int k = (ks + 1) << 5;
      async_b128(sb + aoff, agp + k);
#pragma unroll
      for (int j = 0; j < 4; ++j) async_b128(sb + boff[j], bgp[j] + k);
      asm volatile("s_wait_asynccnt 0x5" ::: "memory");
    } else {
      asm volatile("s_wait_asynccnt 0x0" ::: "memory");
    }
    __syncthreads();

    const bf16* sA = smem + (size_t)cur * STG;
    const bf16* sB = sA + 2048;
    FragBF af[2], bfr[4];
#pragma unroll
    for (int sm = 0; sm < 2; ++sm) {
      const bf16* row = sA + (wm * 32 + sm * 16 + r) * 32;
      af[sm].q[0] = *(const v4u*)(row + half * 8);
      af[sm].q[1] = *(const v4u*)(row + 16 + half * 8);
    }
#pragma unroll
    for (int sn = 0; sn < 4; ++sn) {
      const bf16* row = sB + (wn * 64 + sn * 16 + r) * 32;
      bfr[sn].q[0] = *(const v4u*)(row + half * 8);
      bfr[sn].q[1] = *(const v4u*)(row + 16 + half * 8);
    }
#pragma unroll
    for (int sm = 0; sm < 2; ++sm)
#pragma unroll
      for (int sn = 0; sn < 4; ++sn)
        acc[sm][sn] = __builtin_amdgcn_wmma_f32_16x16x32_bf16(
            false, af[sm].v, false, bfr[sn].v, (short)0, acc[sm][sn], false, false);

    __syncthreads();                               // protect other buffer before next issue
  }

  // epilogue (store guards AFTER all matrix ops: EXEC all-ones during WMMA)
#pragma unroll
  for (int sm = 0; sm < 2; ++sm) {
    int mrow = m0 + wm * 32 + sm * 16 + half * 8;
#pragma unroll
    for (int sn = 0; sn < 4; ++sn) {
      int col = n0 + wn * 64 + sn * 16 + r;
      if (col < Ntotal) {
        float bv = bias ? bias[col] : 0.f;
        float* crow = C + (size_t)mrow * ldc + col;
#pragma unroll
        for (int v = 0; v < 8; ++v) {
          float x = acc[sm][sn][v] + bv;
          if (flags & 1) x = tanhf(x);
          crow[(size_t)v * ldc] = x;
        }
      }
    }
  }
}

// ---------------------------------------------------------------- LSTM gate activations
__global__ void k_lstm_act(const float* __restrict__ g, float* __restrict__ cst,
                           bf16* __restrict__ hst, bf16* __restrict__ out_bf,
                           float* __restrict__ out_f, int t) {
  int i = blockIdx.x * blockDim.x + threadIdx.x;   // over 2*N*DIMH
  if (i >= 2 * NN * DIMH) return;
  int d = i & (DIMH - 1);
  int n = (i >> 9) & (NN - 1);
  int dir = i >> 15;
  const float* gr = g + ((size_t)dir * NN + n) * (4 * DIMH);
  float gi = gr[d], gf = gr[DIMH + d], gg = gr[2 * DIMH + d], go = gr[3 * DIMH + d];
  float c = sigf(gf) * cst[i] + sigf(gi) * tanhf(gg);
  cst[i] = c;
  float h = sigf(go) * tanhf(c);
  hst[i] = f2b(h);
  int time = dir ? (LL - 1 - t) : t;
  size_t o = ((size_t)n * LL + time) * DD + (size_t)dir * DIMH + d;
  out_bf[o] = f2b(h);
  if (out_f) out_f[o] = h;
}

__global__ void k_dec_act(const float* __restrict__ g, bf16* __restrict__ s_b,
                          bf16* __restrict__ outs_b, int step) {
  int i = blockIdx.x * blockDim.x + threadIdx.x;   // over N*DD
  if (i >= NN * DD) return;
  int d = i & (DD - 1), n = i >> 10;
  const float* gr = g + (size_t)n * (4 * DD);
  float gi = gr[d], gg = gr[2 * DD + d], go = gr[3 * DD + d];
  float c = sigf(gi) * tanhf(gg);                  // cell reset each step: f*0 == 0
  float s = sigf(go) * tanhf(c);
  s_b[i] = f2b(s);
  outs_b[((size_t)n * 11 + step) * DD + d] = f2b(s);
}

// ---------------------------------------------------------------- attention
__global__ void k_att_e(const float* __restrict__ Wa_s, const float* __restrict__ Uah,
                        const float* __restrict__ va, const float* __restrict__ va_b,
                        const int* __restrict__ seq_len, float* __restrict__ e) {
  int wave = (int)((blockIdx.x * blockDim.x + threadIdx.x) >> 5);
  int lane = threadIdx.x & 31;
  if (wave >= NN * LL) return;
  int n = wave >> 9;
  int l = wave & (LL - 1);
  const float* u  = Uah  + (size_t)wave * DD;
  const float* ws = Wa_s + (size_t)n * DD;
  float sum = 0.f;
  for (int d = lane; d < DD; d += 32) sum += tanhf(ws[d] + u[d]) * va[d];
  for (int off = 16; off; off >>= 1) sum += __shfl_xor(sum, off, 32);
  if (lane == 0) {
    float v = sum + va_b[0];
    if (l >= seq_len[n]) v += -100000000.0f;
    e[wave] = v;
  }
}

__global__ void k_softmax(float* __restrict__ e) {          // one block per n
  __shared__ float red[8];
  int n = blockIdx.x, tid = threadIdx.x;
  float* row = e + (size_t)n * LL;
  float m = -3.4e38f;
  for (int l = tid; l < LL; l += 256) m = fmaxf(m, row[l]);
  for (int off = 16; off; off >>= 1) m = fmaxf(m, __shfl_xor(m, off, 32));
  if ((tid & 31) == 0) red[tid >> 5] = m;
  __syncthreads();
  float bm = red[0];
#pragma unroll
  for (int j = 1; j < 8; ++j) bm = fmaxf(bm, red[j]);
  __syncthreads();
  float s = 0.f;
  for (int l = tid; l < LL; l += 256) { float v = __expf(row[l] - bm); row[l] = v; s += v; }
  for (int off = 16; off; off >>= 1) s += __shfl_xor(s, off, 32);
  if ((tid & 31) == 0) red[tid >> 5] = s;
  __syncthreads();
  float bs = 0.f;
#pragma unroll
  for (int j = 0; j < 8; ++j) bs += red[j];
  float inv = 1.f / bs;
  for (int l = tid; l < LL; l += 256) row[l] *= inv;
}

__global__ void k_ctx(const float* __restrict__ att, const float* __restrict__ hid,
                      bf16* __restrict__ ci_b) {
  int n = blockIdx.x, tid = threadIdx.x;
  const float* a = att + (size_t)n * LL;
  const float* h = hid + (size_t)n * LL * DD;
  float acc[4] = {0.f, 0.f, 0.f, 0.f};
  for (int l = 0; l < LL; ++l) {
    float w = a[l];
    const float* hr = h + (size_t)l * DD;
#pragma unroll
    for (int j = 0; j < 4; ++j) acc[j] += w * hr[tid + 256 * j];
  }
#pragma unroll
  for (int j = 0; j < 4; ++j) ci_b[(size_t)n * DD + tid + 256 * j] = f2b(acc[j]);
}

// ---------------------------------------------------------------- host
static inline void gemm(hipStream_t s, const bf16* A0, int lda0, int K0,
                        const bf16* A1, int lda1, int K1,
                        const bf16* Bt0, const bf16* Bt1,
                        float* C, int ldc, int Mtiles, int Ntiles,
                        const float* bias, int flags) {
  int total = Mtiles * Ntiles;
  int blocks = (total + 7) / 8;
  k_gemm<<<blocks, 256, 0, s>>>(A0, lda0, K0, A1, lda1, K1, Bt0, Bt1,
                                C, ldc, Ntiles, total, bias, flags);
}
static inline void gemm_lds(hipStream_t s, const bf16* A, int lda, int K,
                            const bf16* Bt, float* C, int ldc,
                            int M, int Ntotal, const float* bias, int flags) {
  int Mblocks = M / 64;
  int Nblocks = (Ntotal + 255) / 256;
  k_gemm_lds<<<Mblocks * Nblocks, 256, 0, s>>>(A, lda, K, Bt, C, ldc,
                                               Nblocks, Ntotal, bias, flags);
}
static inline void cast_launch(hipStream_t s, const void* in, bf16* out, int n) {
  k_cast<<<(n + 255) / 256, 256, 0, s>>>((const float*)in, out, n);
}

extern "C" void kernel_launch(void* const* d_in, const int* in_sizes, int n_in,
                              void* d_out, int out_size, void* d_ws, size_t ws_size,
                              hipStream_t stream) {
  (void)in_sizes; (void)n_in; (void)out_size; (void)ws_size;
  const int*   X       = (const int*)d_in[0];
  const int*   seq_len = (const int*)d_in[1];
  const float* embed_W = (const float*)d_in[2];
  const float* va_W    = (const float*)d_in[25];
  const float* va_b    = (const float*)d_in[26];

  char* p = (char*)d_ws;
  auto alloc = [&](size_t bytes) { void* r = (void*)p; p += (bytes + 255) & ~(size_t)255; return r; };

  bf16*  emb_b  = (bf16*)alloc((size_t)NN * LL * EE * 2);
  bf16*  h0_b   = (bf16*)alloc((size_t)NN * LL * DD * 2);
  bf16*  hid_b  = (bf16*)alloc((size_t)NN * LL * DD * 2);
  float* hid_f  = (float*)alloc((size_t)NN * LL * DD * 4);
  float* uah_f  = (float*)alloc((size_t)NN * LL * DD * 4);

  bf16* w_ih0[2] = {(bf16*)alloc(2048 * 128 * 2),  (bf16*)alloc(2048 * 128 * 2)};
  bf16* w_hh0[2] = {(bf16*)alloc(2048 * 512 * 2),  (bf16*)alloc(2048 * 512 * 2)};
  bf16* w_ih1[2] = {(bf16*)alloc(2048 * 1024 * 2), (bf16*)alloc(2048 * 1024 * 2)};
  bf16* w_hh1[2] = {(bf16*)alloc(2048 * 512 * 2),  (bf16*)alloc(2048 * 512 * 2)};
  bf16* w_Ws  = (bf16*)alloc(1024 * 1024 * 2);
  bf16* w_Wa  = (bf16*)alloc(1024 * 1024 * 2);
  bf16* w_Ua  = (bf16*)alloc(1024 * 1024 * 2);
  bf16* w_dih = (bf16*)alloc(4096 * 1024 * 2);
  bf16* w_dhh = (bf16*)alloc(4096 * 1024 * 2);
  bf16* w_cls = (bf16*)alloc((size_t)10000 * 1024 * 2);

  float* bc0[2] = {(float*)alloc(2048 * 4), (float*)alloc(2048 * 4)};
  float* bc1[2] = {(float*)alloc(2048 * 4), (float*)alloc(2048 * 4)};
  float* bcd    = (float*)alloc(4096 * 4);

  bf16*  hst  = (bf16*)alloc(2 * NN * DIMH * 2);
  float* cst  = (float*)alloc(2 * NN * DIMH * 4);
  float* g_f  = (float*)alloc(2 * NN * 2048 * 4);
  float* s_f  = (float*)alloc(NN * DD * 4);
  bf16*  s_b  = (bf16*)alloc(NN * DD * 2);
  float* was_f = (float*)alloc(NN * DD * 4);
  float* e_f  = (float*)alloc(NN * LL * 4);
  bf16*  ci_b = (bf16*)alloc(NN * DD * 2);
  float* gd_f = (float*)alloc(NN * 4096 * 4);
  bf16*  outs_b = (bf16*)alloc((size_t)NN * 11 * DD * 2);

  // ---- weight conversions (f32 -> bf16; torch (out,in) row-major == B^T layout)
  cast_launch(stream, d_in[3],  w_ih0[0], 2048 * 128);
  cast_launch(stream, d_in[4],  w_hh0[0], 2048 * 512);
  cast_launch(stream, d_in[7],  w_ih0[1], 2048 * 128);
  cast_launch(stream, d_in[8],  w_hh0[1], 2048 * 512);
  cast_launch(stream, d_in[11], w_ih1[0], 2048 * 1024);
  cast_launch(stream, d_in[12], w_hh1[0], 2048 * 512);
  cast_launch(stream, d_in[15], w_ih1[1], 2048 * 1024);
  cast_launch(stream, d_in[16], w_hh1[1], 2048 * 512);
  cast_launch(stream, d_in[19], w_Ws,  1024 * 1024);
  cast_launch(stream, d_in[21], w_Wa,  1024 * 1024);
  cast_launch(stream, d_in[23], w_Ua,  1024 * 1024);
  cast_launch(stream, d_in[27], w_dih, 4096 * 1024);
  cast_launch(stream, d_in[28], w_dhh, 4096 * 1024);
  cast_launch(stream, d_in[31], w_cls, 10000 * 1024);

  k_addb<<<8,  256, 0, stream>>>((const float*)d_in[5],  (const float*)d_in[6],  bc0[0], 2048);
  k_addb<<<8,  256, 0, stream>>>((const float*)d_in[9],  (const float*)d_in[10], bc0[1], 2048);
  k_addb<<<8,  256, 0, stream>>>((const float*)d_in[13], (const float*)d_in[14], bc1[0], 2048);
  k_addb<<<8,  256, 0, stream>>>((const float*)d_in[17], (const float*)d_in[18], bc1[1], 2048);
  k_addb<<<16, 256, 0, stream>>>((const float*)d_in[29], (const float*)d_in[30], bcd,    4096);

  // ---- embedding gather -> bf16
  {
    int total = NN * LL * EE;
    k_embed<<<(total + 255) / 256, 256, 0, stream>>>(X, embed_W, emb_b, total);
  }

  // ---- encoder: 2 layers, fwd+bwd; recurrence fused with input projection
  for (int layer = 0; layer < 2; ++layer) {
    k_zero<<<(2 * NN * DIMH / 2 + 255) / 256, 256, 0, stream>>>((float*)hst, 2 * NN * DIMH / 2);
    k_zero<<<(2 * NN * DIMH + 255) / 256, 256, 0, stream>>>(cst, 2 * NN * DIMH);
    const bf16* xin  = layer ? h0_b : emb_b;
    int         Kin  = layer ? DD : EE;
    bf16*       outb = layer ? hid_b : h0_b;
    float*      outf = layer ? hid_f : nullptr;
    for (int t = 0; t < LL; ++t) {
      for (int dir = 0; dir < 2; ++dir) {
        int time = dir ? (LL - 1 - t) : t;
        // g = h @ Whh^T + x_t @ Wih^T + (bih+bhh)   -> (64 x 2048)
        gemm(stream,
             hst + (size_t)dir * NN * DIMH, DIMH, DIMH,
             xin + (size_t)time * Kin, LL * Kin, Kin,
             layer ? w_hh1[dir] : w_hh0[dir],
             layer ? w_ih1[dir] : w_ih0[dir],
             g_f + (size_t)dir * NN * 2048, 2048,
             NN / 16, 2048 / 16,
             layer ? bc1[dir] : bc0[dir], 0);
      }
      k_lstm_act<<<(2 * NN * DIMH + 255) / 256, 256, 0, stream>>>(g_f, cst, hst, outb, outf, t);
    }
  }

  // ---- hoisted Ua_h = hid @ Ua^T + Ua_b  (32768 x 1024 x 1024) — LDS/async kernel
  gemm_lds(stream, hid_b, DD, DD, w_Ua, uah_f, DD,
           NN * LL, DD, (const float*)d_in[24], 0);

  // ---- s_0 = tanh(hid[:,0] @ Ws^T + Ws_b)
  gemm(stream, hid_b, LL * DD, DD, nullptr, 0, 0, w_Ws, nullptr,
       s_f, DD, NN / 16, DD / 16, (const float*)d_in[20], 1);
  cast_launch(stream, s_f, s_b, NN * DD);

  // ---- attention decoder: 11 steps
  for (int step = 0; step < 11; ++step) {
    gemm(stream, s_b, DD, DD, nullptr, 0, 0, w_Wa, nullptr,
         was_f, DD, NN / 16, DD / 16, (const float*)d_in[22], 0);
    k_att_e<<<(NN * LL * 32 + 255) / 256, 256, 0, stream>>>(was_f, uah_f, va_W, va_b, seq_len, e_f);
    k_softmax<<<NN, 256, 0, stream>>>(e_f);
    k_ctx<<<NN, 256, 0, stream>>>(e_f, hid_f, ci_b);
    gemm(stream, ci_b, DD, DD, s_b, DD, DD, w_dih, w_dhh,
         gd_f, 4096, NN / 16, 4096 / 16, bcd, 0);
    k_dec_act<<<(NN * DD + 255) / 256, 256, 0, stream>>>(gd_f, s_b, outs_b, step);
  }

  // ---- classifier: (704 x 10000 x 1024) — LDS/async kernel, N guarded
  gemm_lds(stream, outs_b, DD, DD, w_cls, (float*)d_out, 10000,
           NN * 11, 10000, (const float*)d_in[32], 0);
}
